// SequenceEmbedding_Attention_89137751261308
// MI455X (gfx1250) — matrix operand order repaired and checked by reading, not compiled, not run on hardware
//
#include <hip/hip_runtime.h>
#include <hip/hip_bf16.h>
#include <math.h>

// Problem constants (from reference setup_inputs)
#define B_ 32
#define S_ 2048
#define H_ 2048
#define Q_ 1024
#define MIN_NUMBER -1e32f

typedef __attribute__((ext_vector_type(16))) __bf16 v16bf;
typedef __attribute__((ext_vector_type(8)))  __bf16 v8bf;
typedef __attribute__((ext_vector_type(8)))  float  v8f;

// Branchless tanh: exact at the saturation ends (expf->inf => 1, expf->0 => -1)
__device__ __forceinline__ float fast_tanh(float x) {
    float ex = __expf(2.0f * x);
    return 1.0f - 2.0f / (ex + 1.0f);
}

// ---------------------------------------------------------------------------
// A-matrix tile (16x32 bf16) from bf16 memory: lane row m = lane&15.
// Elements <-> K = 16*half + i (one contiguous 32-byte chunk).
// ---------------------------------------------------------------------------
__device__ __forceinline__ v16bf load_tile_A_bf(const __bf16* __restrict__ p, int half) {
    v8bf lo = *(const v8bf*)(p + (half << 4));
    v8bf hi = *(const v8bf*)(p + (half << 4) + 8);
    return __builtin_shufflevector(lo, hi, 0,1,2,3,4,5,6,7,8,9,10,11,12,13,14,15);
}

// ---------------------------------------------------------------------------
// B-matrix tile (32x16 K x N) from bf16 memory: lane col n = lane&15,
// elements <-> K = 16*half + i (one contiguous 32-byte chunk).
// ---------------------------------------------------------------------------
__device__ __forceinline__ v16bf load_tile_B_bf(const __bf16* __restrict__ p, int half) {
    v8bf lo = *(const v8bf*)(p + (half << 4));
    v8bf hi = *(const v8bf*)(p + (half << 4) + 8);
    return __builtin_shufflevector(lo, hi, 0,1,2,3,4,5,6,7,8,9,10,11,12,13,14,15);
}

// ---------------------------------------------------------------------------
// B-matrix tile from fp32 with on-the-fly bf16 convert (fallback path).
// ---------------------------------------------------------------------------
__device__ __forceinline__ v16bf load_tile_B_f32(const float* __restrict__ p, int half) {
    const float* q = p + (half << 4);
    float4 x0 = *(const float4*)(q);
    float4 x1 = *(const float4*)(q + 4);
    float4 x2 = *(const float4*)(q + 8);
    float4 x3 = *(const float4*)(q + 12);
    v16bf r;
    r[0]  = (__bf16)x0.x; r[1]  = (__bf16)x0.y; r[2]  = (__bf16)x0.z; r[3]  = (__bf16)x0.w;
    r[4]  = (__bf16)x1.x; r[5]  = (__bf16)x1.y; r[6]  = (__bf16)x1.z; r[7]  = (__bf16)x1.w;
    r[8]  = (__bf16)x2.x; r[9]  = (__bf16)x2.y; r[10] = (__bf16)x2.z; r[11] = (__bf16)x2.w;
    r[12] = (__bf16)x3.x; r[13] = (__bf16)x3.y; r[14] = (__bf16)x3.z; r[15] = (__bf16)x3.w;
    return r;
}

// ---------------------------------------------------------------------------
// fp32 -> bf16 bulk convert (8 elements/thread, b128 in / b128 out)
// ---------------------------------------------------------------------------
__global__ void cvt_bf16_kernel(const float* __restrict__ src, __bf16* __restrict__ dst) {
    const size_t i = ((size_t)blockIdx.x * blockDim.x + threadIdx.x) * 8;
    float4 x0 = *(const float4*)(src + i);
    float4 x1 = *(const float4*)(src + i + 4);
    v8bf o;
    o[0] = (__bf16)x0.x; o[1] = (__bf16)x0.y; o[2] = (__bf16)x0.z; o[3] = (__bf16)x0.w;
    o[4] = (__bf16)x1.x; o[5] = (__bf16)x1.y; o[6] = (__bf16)x1.z; o[7] = (__bf16)x1.w;
    *(v8bf*)(dst + i) = o;
}

// ---------------------------------------------------------------------------
// c1[b,q] = sum_p rth[b,p] * Wa[q,p]   (tiny, 67 MFLOP)
// ---------------------------------------------------------------------------
__global__ void c1_kernel(const float* __restrict__ rth, const float* __restrict__ Wa,
                          float* __restrict__ c1) {
    const int idx = blockIdx.x * blockDim.x + threadIdx.x;   // b*Q + q
    const int b = idx >> 10;
    const int q = idx & (Q_ - 1);
    const float4* wr = (const float4*)(Wa + (size_t)q * Q_);
    const float4* xr = (const float4*)(rth + (size_t)b * Q_);
    float acc = 0.f;
    #pragma unroll 4
    for (int p = 0; p < Q_ / 4; ++p) {
        float4 w = wr[p], x = xr[p];
        acc += w.x * x.x + w.y * x.y + w.z * x.z + w.w * x.w;
    }
    c1[idx] = acc;
}

// ---------------------------------------------------------------------------
// HOT kernel, fast path: 4q x 4s register blocking, both operands bf16.
// One wave owns a 64-wide s-strip of batch b; 16 accumulators (64x64 out).
// Per 32-deep K step: 8 A loads + 8 B loads (b128, zero cvt) -> 16 WMMAs.
// Epilogue fuses Va[q]*tanh(c1+c2), reduces over q; plain stores, no atomics.
// ---------------------------------------------------------------------------
#define SCORE_WAVES 4
__global__ void __launch_bounds__(32 * SCORE_WAVES)
attn_score_bf_kernel(const __bf16* __restrict__ rnnbf, const __bf16* __restrict__ Uabf,
                     const float* __restrict__ c1, const float* __restrict__ Va,
                     float* __restrict__ e) {
    const int lane = threadIdx.x & 31;
    const int wave = threadIdx.x >> 5;
    const int half = lane >> 4;
    const int lm   = lane & 15;
    const int b    = blockIdx.y;
    const int s0   = (blockIdx.x * SCORE_WAVES + wave) * 64;   // 64-wide strip

    const __bf16* rnnRow[4];
    #pragma unroll
    for (int u = 0; u < 4; ++u)
        rnnRow[u] = rnnbf + ((size_t)b * S_ + (size_t)(s0 + 16 * u + lm)) * H_;
    const float* c1b = c1 + (size_t)b * Q_;

    float esum[4] = {0.f, 0.f, 0.f, 0.f};

    for (int q0 = 0; q0 < Q_; q0 += 64) {
        const __bf16* uaRow[4];
        #pragma unroll
        for (int t = 0; t < 4; ++t)
            uaRow[t] = Uabf + (size_t)(q0 + 16 * t + lm) * H_;

        v8f acc[4][4];
        #pragma unroll
        for (int t = 0; t < 4; ++t)
            #pragma unroll
            for (int u = 0; u < 4; ++u)
                acc[t][u] = (v8f){};

        for (int h0 = 0; h0 < H_; h0 += 32) {
            v16bf am[4], bm[4];
            #pragma unroll
            for (int t = 0; t < 4; ++t) am[t] = load_tile_A_bf(uaRow[t] + h0, half);
            #pragma unroll
            for (int u = 0; u < 4; ++u) bm[u] = load_tile_B_bf(rnnRow[u] + h0, half);
            #pragma unroll
            for (int t = 0; t < 4; ++t)
                #pragma unroll
                for (int u = 0; u < 4; ++u)
                    acc[t][u] = __builtin_amdgcn_wmma_f32_16x16x32_bf16(
                        false, am[t], false, bm[u], (short)0, acc[t][u], false, false);
        }

        // D element r <-> (q = q0 + 16t + r + 8*half, s = s0 + 16u + lm)
        #pragma unroll
        for (int t = 0; t < 4; ++t) {
            #pragma unroll
            for (int r = 0; r < 8; ++r) {
                const int q  = q0 + 16 * t + r + 8 * half;
                const float cq = c1b[q];
                const float vq = Va[q];
                #pragma unroll
                for (int u = 0; u < 4; ++u)
                    esum[u] += vq * fast_tanh(cq + acc[t][u][r]);
            }
        }
    }

    #pragma unroll
    for (int u = 0; u < 4; ++u) esum[u] += __shfl_xor(esum[u], 16, 32);
    if (lane < 16) {
        #pragma unroll
        for (int u = 0; u < 4; ++u)
            e[(size_t)b * S_ + s0 + 16 * u + lane] = esum[u];
    }
}

// ---------------------------------------------------------------------------
// HOT kernel, fallback path (small ws): 2q x 2s, B converted fp32->bf16 in-loop.
// ---------------------------------------------------------------------------
__global__ void __launch_bounds__(32 * SCORE_WAVES)
attn_score_f32_kernel(const float* __restrict__ rnn, const __bf16* __restrict__ Uabf,
                      const float* __restrict__ c1, const float* __restrict__ Va,
                      float* __restrict__ e) {
    const int lane = threadIdx.x & 31;
    const int wave = threadIdx.x >> 5;
    const int half = lane >> 4;
    const int lm   = lane & 15;
    const int b    = blockIdx.y;
    const int s0   = (blockIdx.x * SCORE_WAVES + wave) * 32;

    const float* rnnRow0 = rnn + ((size_t)b * S_ + (size_t)(s0 + lm)) * H_;
    const float* rnnRow1 = rnnRow0 + (size_t)16 * H_;
    const float* c1b     = c1 + (size_t)b * Q_;

    float esum0 = 0.f, esum1 = 0.f;

    for (int q0 = 0; q0 < Q_; q0 += 32) {
        const __bf16* uaRow0 = Uabf + (size_t)(q0 + lm) * H_;
        const __bf16* uaRow1 = uaRow0 + (size_t)16 * H_;

        v8f acc00 = {}, acc01 = {}, acc10 = {}, acc11 = {};
        #pragma unroll 2
        for (int h0 = 0; h0 < H_; h0 += 32) {
            v16bf a0 = load_tile_A_bf(uaRow0 + h0, half);
            v16bf a1 = load_tile_A_bf(uaRow1 + h0, half);
            v16bf b0 = load_tile_B_f32(rnnRow0 + h0, half);
            v16bf b1 = load_tile_B_f32(rnnRow1 + h0, half);
            acc00 = __builtin_amdgcn_wmma_f32_16x16x32_bf16(false, a0, false, b0, (short)0, acc00, false, false);
            acc01 = __builtin_amdgcn_wmma_f32_16x16x32_bf16(false, a0, false, b1, (short)0, acc01, false, false);
            acc10 = __builtin_amdgcn_wmma_f32_16x16x32_bf16(false, a1, false, b0, (short)0, acc10, false, false);
            acc11 = __builtin_amdgcn_wmma_f32_16x16x32_bf16(false, a1, false, b1, (short)0, acc11, false, false);
        }

        #pragma unroll
        for (int r = 0; r < 8; ++r) {
            const int qa = q0 + r + 8 * half;
            const int qb = qa + 16;
            const float ca = c1b[qa], cb = c1b[qb];
            const float va = Va[qa],  vb = Va[qb];
            esum0 += va * fast_tanh(ca + acc00[r]);
            esum1 += va * fast_tanh(ca + acc01[r]);
            esum0 += vb * fast_tanh(cb + acc10[r]);
            esum1 += vb * fast_tanh(cb + acc11[r]);
        }
    }

    esum0 += __shfl_xor(esum0, 16, 32);
    esum1 += __shfl_xor(esum1, 16, 32);
    if (lane < 16) {
        e[(size_t)b * S_ + s0 + lane]      = esum0;
        e[(size_t)b * S_ + s0 + 16 + lane] = esum1;
    }
}

// ---------------------------------------------------------------------------
// Masked softmax over S per batch row.
// ---------------------------------------------------------------------------
__global__ void softmax_kernel(const float* __restrict__ e, const int* __restrict__ mask,
                               float* __restrict__ a) {
    __shared__ float sdata[256];
    const int b = blockIdx.x, tid = threadIdx.x;
    float vals[S_ / 256];
    float mx = -3.402823466e38f;
    #pragma unroll
    for (int i = 0; i < S_ / 256; ++i) {
        const int s = i * 256 + tid;
        float v = e[(size_t)b * S_ + s];
        if (mask[(size_t)b * S_ + s] == 0) v = MIN_NUMBER;
        vals[i] = v;
        mx = fmaxf(mx, v);
    }
    sdata[tid] = mx; __syncthreads();
    for (int off = 128; off > 0; off >>= 1) {
        if (tid < off) sdata[tid] = fmaxf(sdata[tid], sdata[tid + off]);
        __syncthreads();
    }
    mx = sdata[0]; __syncthreads();

    float sum = 0.f;
    #pragma unroll
    for (int i = 0; i < S_ / 256; ++i) {
        vals[i] = __expf(vals[i] - mx);
        sum += vals[i];
    }
    sdata[tid] = sum; __syncthreads();
    for (int off = 128; off > 0; off >>= 1) {
        if (tid < off) sdata[tid] += sdata[tid + off];
        __syncthreads();
    }
    const float inv = 1.0f / sdata[0];
    #pragma unroll
    for (int i = 0; i < S_ / 256; ++i) {
        const int s = i * 256 + tid;
        a[(size_t)b * S_ + s] = vals[i] * inv;
    }
}

// ---------------------------------------------------------------------------
// context[b,h] = sum_s a[b,s] * rnn[b,s,h]  (fp32 streaming pass)
// ---------------------------------------------------------------------------
__global__ void context_kernel(const float* __restrict__ rnn, const float* __restrict__ a,
                               float* __restrict__ out) {
    const int b = blockIdx.y;
    const int h = blockIdx.x * blockDim.x + threadIdx.x;
    const float* rb = rnn + (size_t)b * S_ * H_ + h;
    const float* ab = a + (size_t)b * S_;
    float a0 = 0.f, a1 = 0.f, a2 = 0.f, a3 = 0.f;
    for (int s = 0; s < S_; s += 4) {
        a0 += ab[s + 0] * rb[(size_t)(s + 0) * H_];
        a1 += ab[s + 1] * rb[(size_t)(s + 1) * H_];
        a2 += ab[s + 2] * rb[(size_t)(s + 2) * H_];
        a3 += ab[s + 3] * rb[(size_t)(s + 3) * H_];
    }
    out[(size_t)b * H_ + h] = (a0 + a1) + (a2 + a3);
}

// ---------------------------------------------------------------------------
extern "C" void kernel_launch(void* const* d_in, const int* in_sizes, int n_in,
                              void* d_out, int out_size, void* d_ws, size_t ws_size,
                              hipStream_t stream) {
    const float* rnn  = (const float*)d_in[0];   // [B,S,H]
    const float* rth  = (const float*)d_in[1];   // [B,Q]
    const int*   mask = (const int*)  d_in[2];   // [B,S]
    const float* Wa   = (const float*)d_in[3];   // [Q,Q]
    const float* Ua   = (const float*)d_in[4];   // [Q,H]
    const float* Va   = (const float*)d_in[5];   // [Q]
    float* out = (float*)d_out;                  // [B,H]

    float*  c1    = (float*)d_ws;                     // B*Q  floats
    float*  e     = c1 + (size_t)B_ * Q_;             // B*S  floats
    float*  a     = e  + (size_t)B_ * S_;             // B*S  floats
    __bf16* Uabf  = (__bf16*)(a + (size_t)B_ * S_);   // Q*H bf16  (4 MB)
    __bf16* rnnbf = Uabf + (size_t)Q_ * H_;           // B*S*H bf16 (256 MB, optional)

    const size_t small_bytes = ((size_t)B_ * Q_ + 2 * (size_t)B_ * S_) * 4 + (size_t)Q_ * H_ * 2;
    const size_t big_bytes   = small_bytes + (size_t)B_ * S_ * H_ * 2;

    // 0) Ua -> bf16 (always; 4MB, L2-resident afterwards)
    cvt_bf16_kernel<<<(Q_ * H_) / (256 * 8), 256, 0, stream>>>(Ua, Uabf);

    // 1) c1 = rth @ Wa.T
    c1_kernel<<<(B_ * Q_) / 256, 256, 0, stream>>>(rth, Wa, c1);

    // 2) fused WMMA scores
    if (ws_size >= big_bytes) {
        // one-time rnn -> bf16 (halves GEMM bytes, kills all in-loop cvt)
        cvt_bf16_kernel<<<(int)(((size_t)B_ * S_ * H_) / (256 * 8)), 256, 0, stream>>>(rnn, rnnbf);
        dim3 sgrid(S_ / 64 / SCORE_WAVES, B_);        // (8, 32) blocks of 4 waves
        attn_score_bf_kernel<<<sgrid, 32 * SCORE_WAVES, 0, stream>>>(rnnbf, Uabf, c1, Va, e);
    } else {
        dim3 sgrid(S_ / 32 / SCORE_WAVES, B_);        // (16, 32) blocks of 4 waves
        attn_score_f32_kernel<<<sgrid, 32 * SCORE_WAVES, 0, stream>>>(rnn, Uabf, c1, Va, e);
    }

    // 3) masked softmax over S
    softmax_kernel<<<B_, 256, 0, stream>>>(e, mask, a);

    // 4) context = sum_s a[b,s] * rnn[b,s,:]
    dim3 cgrid(H_ / 256, B_);
    context_kernel<<<cgrid, 256, 0, stream>>>(rnn, a, out);
}
